// MoD_90263032692829
// MI455X (gfx1250) — matrix operand order repaired and checked by reading, not compiled, not run on hardware
//
#include <hip/hip_runtime.h>

// ---------------- problem constants ----------------
#define AA    8
#define CH    256
#define SS    16384     // S1*D1
#define KSEL  2048
#define NSEL  2047
#define MTOT  2048      // L = s1 * c
#define NTAP  9

typedef __attribute__((ext_vector_type(16))) __bf16 v16bf;
typedef __attribute__((ext_vector_type(8)))  float  v8f;

#define GLOBAL_AS __attribute__((address_space(1)))
#define LDS_AS    __attribute__((address_space(3)))

#if defined(__has_builtin)
#  if __has_builtin(__builtin_amdgcn_global_load_async_to_lds_b32) && \
      __has_builtin(__builtin_amdgcn_s_wait_asynccnt)
#    define USE_ASYNC 1
#  endif
#endif
#ifndef USE_ASYNC
#  define USE_ASYNC 0
#endif

// ---------------- helpers ----------------
__device__ __forceinline__ unsigned short f2bf(float f) {
    union { float f; unsigned int u; } v; v.f = f;
    unsigned int u = v.u;
    unsigned int r = (u + 0x7fffu + ((u >> 16) & 1u)) >> 16;   // RNE
    return (unsigned short)r;
}
__device__ __forceinline__ unsigned int f2ord(float f) {
    union { float f; unsigned int u; } v; v.f = f;
    return (v.u & 0x80000000u) ? ~v.u : (v.u | 0x80000000u);
}
__device__ __forceinline__ float ord2f(unsigned int o) {
    union { float f; unsigned int u; } v;
    v.u = (o & 0x80000000u) ? (o & 0x7fffffffu) : ~o;
    return v.f;
}

// ---------------- 1. router ----------------
__global__ __launch_bounds__(256) void router_kernel(
    const float* __restrict__ x, const float* __restrict__ rw,
    const float* __restrict__ rb, float* __restrict__ wts)
{
    int i   = blockIdx.x * 256 + threadIdx.x;        // 8*16384
    int pos = i & (SS - 1);
    int a   = i >> 14;
    const float* xa = x + ((size_t)a << 22) + pos;
    float sum = 0.f;
    #pragma unroll 4
    for (int c = 0; c < CH; ++c) sum += xa[(size_t)c << 14] * rw[c];
    wts[i] = sum + rb[0];
}

// ---------------- 2. exact k-th-largest per row via 4-pass radix select ----------------
__global__ __launch_bounds__(256) void topk_thresh_kernel(
    const float* __restrict__ wts, float* __restrict__ thr)
{
    int a = blockIdx.x, tid = threadIdx.x;
    __shared__ unsigned int hist[256];
    __shared__ unsigned int s_prefix, s_rank;
    if (tid == 0) { s_prefix = 0u; s_rank = (unsigned)KSEL; }
    __syncthreads();
    const float* w = wts + ((size_t)a << 14);
    for (int pass = 0; pass < 4; ++pass) {
        int shift = 24 - 8 * pass;
        hist[tid] = 0u;
        __syncthreads();
        unsigned int pfx = s_prefix;
        for (int i = tid; i < SS; i += 256) {
            unsigned int u = f2ord(w[i]);
            bool match = (pass == 0) || (((u ^ pfx) >> (shift + 8)) == 0u);
            if (match) atomicAdd(&hist[(u >> shift) & 255u], 1u);
        }
        __syncthreads();
        if (tid == 0) {
            unsigned int rank = s_rank, cum = 0u;
            for (int b = 255; b >= 0; --b) {
                unsigned int h = hist[b];
                if (cum + h >= rank) {
                    s_prefix = pfx | ((unsigned int)b << shift);
                    s_rank   = rank - cum;
                    break;
                }
                cum += h;
            }
        }
        __syncthreads();
    }
    if (tid == 0) thr[a] = ord2f(s_prefix);
}

// ---------------- 3. stable compaction of selected positions ----------------
__global__ __launch_bounds__(1024) void compact_kernel(
    const float* __restrict__ wts, const float* __restrict__ thr,
    int* __restrict__ idx)
{
    int a = blockIdx.x, tid = threadIdx.x;
    float t = thr[a];
    const float* w = wts + ((size_t)a << 14);
    unsigned int m16 = 0u;
    #pragma unroll
    for (int j = 0; j < 16; ++j)
        if (w[tid * 16 + j] > t) m16 |= (1u << j);
    unsigned int cnt = __popc(m16);

    __shared__ unsigned int sc[1024];
    sc[tid] = cnt;
    __syncthreads();
    for (int off = 1; off < 1024; off <<= 1) {
        unsigned int v = (tid >= off) ? sc[tid - off] : 0u;
        __syncthreads();
        sc[tid] += v;
        __syncthreads();
    }
    unsigned int run = sc[tid] - cnt;                // exclusive prefix
    #pragma unroll
    for (int j = 0; j < 16; ++j) {
        if (m16 & (1u << j)) {
            if (run < (unsigned)NSEL) idx[a * MTOT + run] = tid * 16 + j;
            ++run;
        }
    }
}

// ---------------- 4. gather selected tokens -> bf16 blk[a][ci][slot] ----------------
__global__ __launch_bounds__(256) void gather_kernel(
    const float* __restrict__ x, const int* __restrict__ idx,
    unsigned short* __restrict__ blk)
{
    int i    = blockIdx.x * 256 + threadIdx.x;       // 8*256*2048
    int slot = i & (MTOT - 1);
    int ci   = (i >> 11) & 255;
    int a    = i >> 19;
    unsigned short v = 0;
    if (slot < NSEL) {
        int pos = idx[a * MTOT + slot] & (SS - 1);
        v = f2bf(x[((size_t)(a * CH + ci) << 14) + pos]);
    }
    blk[i] = v;
}

// ---------------- 5. weights -> bf16 wt[tap][ci/2][co][ci&1] (dword = ci-pair) ----------------
__global__ __launch_bounds__(256) void wprep_kernel(
    const float* __restrict__ bw, unsigned short* __restrict__ wt)
{
    int i   = blockIdx.x * 256 + threadIdx.x;        // 9*128*256*2 = 589824
    int pl  = i & 1;
    int co  = (i >> 1) & 255;
    int cip = (i >> 9) & 127;
    int tap = i >> 16;
    int ci  = cip * 2 + pl;
    wt[i] = f2bf(bw[(co * CH + ci) * NTAP + tap]);
}

// ---------------- 6. passthrough: out = mask ? 0 : x ----------------
__global__ __launch_bounds__(256) void passthrough_kernel(
    const float* __restrict__ x, const float* __restrict__ wts,
    const float* __restrict__ thr, float* __restrict__ out)
{
    int i   = blockIdx.x * 256 + threadIdx.x;        // 2^25 exact
    int pos = i & (SS - 1);
    int a   = i >> 22;
    float w = wts[(a << 14) + pos];
    out[i]  = (w > thr[a]) ? 0.f : x[i];
}

// ---------------- 7. implicit-GEMM 3x3 conv via v_wmma_f32_16x16x32_bf16 ----------------
// block: 256 thr (8 waves), tile 64 M x 128 N, K-chunk 64, double-buffered LDS.
// wave = 16 M x 64 N (4 accumulators); 8 WMMAs per wave per K-iteration.
__global__ __launch_bounds__(256) void conv_wmma_kernel(
    const unsigned short* __restrict__ blk,  // [8][256][2048] bf16
    const unsigned short* __restrict__ wt,   // [9][128][256][2] bf16 (ci-pair packed)
    const float* __restrict__ bias,          // [256]
    const int* __restrict__ idx,             // [8][2048]
    float* __restrict__ out)                 // [8][256][16384]
{
    const int a    = blockIdx.z;
    const int m0   = blockIdx.x * 64;
    const int n0   = blockIdx.y * 128;
    const int tid  = threadIdx.x;
    const int lane = tid & 31;
    const int wave = tid >> 5;
    const int wm   = (wave & 3) * 16;      // wave M offset
    const int wng  = (wave >> 2) * 64;     // wave N group offset (0 or 64)
    const int g    = lane >> 4;            // lane half-group
    const int lr   = lane & 15;

    __shared__ unsigned short lA[2][64 * 68];   // [m][ci]  pitch 68 halves (34 dw)
    __shared__ unsigned int   lB[2][128 * 33];  // [co][ci-pair] pitch 33 dwords

    const unsigned int* wtdw = (const unsigned int*)wt;

    v8f acc[4] = {{}, {}, {}, {}};

    auto stage = [&](int it, int bi) {
        const int tap = it >> 2;               // 4 ci-chunks per tap
        const int c00 = (it & 3) * 64;
        const int kh  = tap / 3 - 1;
        const int kw  = tap % 3 - 1;
        // --- A tile: 64 slots x 64 cin, pad-aware gather ---
        #pragma unroll
        for (int i = 0; i < 16; ++i) {
            int e  = tid + i * 256;
            int m  = e & 63;
            int cl = e >> 6;
            int slot = m0 + m;
            int hh = (slot >> 4) + kh;
            int ww = (slot & 15) + kw;
            unsigned short v = 0;
            if ((unsigned)hh < 128u && (unsigned)ww < 16u)
                v = blk[((size_t)(a * CH + c00 + cl) << 11) + (hh << 4) + ww];
            lA[bi][m * 68 + cl] = v;
        }
        // --- B tile: 32 ci-pairs x 128 co dwords, transposed landing [co][cip] ---
        #pragma unroll
        for (int i = 0; i < 16; ++i) {
            int e   = tid + i * 256;
            int co  = e & 127;
            int cip = e >> 7;                  // 0..31
            int gidx = (tap * (CH / 2) + (c00 >> 1) + cip) * CH + n0 + co;
#if USE_ASYNC
            __builtin_amdgcn_global_load_async_to_lds_b32(
                (GLOBAL_AS int*)(wtdw + gidx),
                (LDS_AS int*)&lB[bi][co * 33 + cip], 0, 0);
#else
            lB[bi][co * 33 + cip] = wtdw[gidx];
#endif
        }
    };

    auto compute = [&](int bi) {
        const unsigned int* Adw = (const unsigned int*)&lA[bi][0];
        #pragma unroll
        for (int s = 0; s < 2; ++s) {
            union { v16bf v; unsigned int u[8]; } fa;
            #pragma unroll
            for (int vv = 0; vv < 8; ++vv) {
                int d = ((vv < 4) ? vv : (8 + vv - 4)) + 4 * g + 16 * s;
                fa.u[vv] = Adw[(wm + lr) * 34 + d];
            }
            #pragma unroll
            for (int j = 0; j < 4; ++j) {
                union { v16bf v; unsigned int u[8]; } fb;
                #pragma unroll
                for (int vv = 0; vv < 8; ++vv)
                    fb.u[vv] = lB[bi][(wng + j * 16 + lr) * 33 + s * 16 + 4 * g + ((vv < 4) ? vv : (vv + 4))];
                acc[j] = __builtin_amdgcn_wmma_f32_16x16x32_bf16(
                    false, fa.v, false, fb.v, (short)0, acc[j], false, false);
            }
        }
    };

    stage(0, 0);
#if USE_ASYNC
    __builtin_amdgcn_s_wait_asynccnt(0);
#endif
    __syncthreads();
    for (int it = 0; it < 36; ++it) {
        int cur = it & 1;
        if (it + 1 < 36) stage(it + 1, cur ^ 1);
        compute(cur);
#if USE_ASYNC
        __builtin_amdgcn_s_wait_asynccnt(0);
#endif
        __syncthreads();
    }

    // bias + scatter through idx (pad slot 2047 dropped)
    #pragma unroll
    for (int j = 0; j < 4; ++j) {
        int co = n0 + wng + j * 16 + lr;
        float b = bias[co];
        #pragma unroll
        for (int r = 0; r < 8; ++r) {
            int slot = m0 + wm + g * 8 + r;        // C/D layout: M = r + 8*lanegroup
            if (slot < NSEL) {
                int pos = idx[a * MTOT + slot] & (SS - 1);
                out[((size_t)(a * CH + co) << 14) + pos] = acc[j][r] + b;
            }
        }
    }
}

// ---------------- launch ----------------
extern "C" void kernel_launch(void* const* d_in, const int* in_sizes, int n_in,
                              void* d_out, int out_size, void* d_ws, size_t ws_size,
                              hipStream_t stream) {
    const float* x  = (const float*)d_in[0];
    const float* rw = (const float*)d_in[1];
    const float* rb = (const float*)d_in[2];
    const float* bw = (const float*)d_in[3];
    const float* bb = (const float*)d_in[4];
    float* out = (float*)d_out;

    char* ws = (char*)d_ws;
    float*          wts = (float*)(ws);                          // 8*16384*4      = 524288
    float*          thr = (float*)(ws + 524288);                 // 64 B slot
    int*            idx = (int*)(ws + 524352);                   // 8*2048*4       = 65536
    unsigned short* blk = (unsigned short*)(ws + 589888);        // 8*256*2048*2   = 8388608
    unsigned short* wtb = (unsigned short*)(ws + 8978496);       // 9*128*256*2*2  = 1179648

    router_kernel     <<<512,    256, 0, stream>>>(x, rw, rb, wts);
    topk_thresh_kernel<<<AA,     256, 0, stream>>>(wts, thr);
    compact_kernel    <<<AA,    1024, 0, stream>>>(wts, thr, idx);
    gather_kernel     <<<16384,  256, 0, stream>>>(x, idx, blk);
    wprep_kernel      <<<2304,   256, 0, stream>>>(bw, wtb);
    passthrough_kernel<<<131072, 256, 0, stream>>>(x, wts, thr, out);

    dim3 grid(MTOT / 64, CH / 128, AA);   // 32 x 2 x 8
    conv_wmma_kernel  <<<grid,   256, 0, stream>>>(blk, wtb, bb, idx, out);
}
